// MultiheadAttentionNLP_20246475834056
// MI455X (gfx1250) — matrix operand order repaired and checked by reading, not compiled, not run on hardware
//
#include <hip/hip_runtime.h>
#include <hip/hip_bf16.h>
#include <stdint.h>

// ---------------------------------------------------------------------------
// MultiheadAttentionNLP on MI455X (gfx1250): f16 WMMA pipeline, f32 accumulate
//   B=2 S=2048 H=16 DH=64 D=1024 OUT=1024
//   - WMMA 16x16x32 f16 everywhere (projections, QK^T, PV, output proj)
//   - V projection writes V^T [b][h][dh][s] so PV fragments are contiguous
//   - K/V^T staged to LDS with gfx1250 async-to-LDS ops, double buffered
//   - 4 waves/block share one (b,h) K/V stream in attention
//   - GEMM k-loop: explicit 2-stage register double buffering (no rotation)
// ---------------------------------------------------------------------------

typedef __attribute__((ext_vector_type(8)))  _Float16 h8;
typedef __attribute__((ext_vector_type(16))) _Float16 v16h;
typedef __attribute__((ext_vector_type(8)))  float    v8f;

#define BB   2
#define SS   2048
#define HHD  16
#define DHH  64
#define DD   1024
#define MM   (BB * SS)       // 4096 rows
#define NEGBIG (-1.0e9f)
#define NT   (SS / 32)       // 64 key tiles of 32

// ---- WMMA fragment helpers (contiguous k order k = 16*(lane/16) + e, wave32) ----------

__device__ __forceinline__ v16h make16(h8 lo, h8 hi) {
  v16h r;
#pragma unroll
  for (int i = 0; i < 8; ++i) { r[i] = lo[i]; r[8 + i] = hi[i]; }
  return r;
}

// A fragment, 16x32 f16 tile of row-major [*, ld] starting at p (=&A[row0,k0]).
// lanes 0-15: row=lane,    e0..15 = K0..15
// lanes16-31: row=lane-16, e0..15 = K16..31
__device__ __forceinline__ v16h loadA(const _Float16* p, int ld, int lane) {
  int row = lane & 15;
  int off = (lane & 16) ? 16 : 0;
  const _Float16* r = p + (size_t)row * ld + off;
  return *(const v16h*)r;
}

// B fragment, 32x16 f16, from "transposed" storage pc[col][k] (row-major ld):
// lanes 0-15: col=lane,    e0..15 = K0..15
// lanes16-31: col=lane-16, e0..15 = K16..31
__device__ __forceinline__ v16h loadBT(const _Float16* pc, int ld, int lane) {
  int col = lane & 15;
  int off = (lane & 16) ? 16 : 0;
  return *(const v16h*)(pc + (size_t)col * ld + off);
}

__device__ __forceinline__ v8f wmma_f16(v16h a, v16h b, v8f c) {
  return __builtin_amdgcn_wmma_f32_16x16x32_f16(false, a, false, b,
                                                (short)0, c, false, false);
}

// ---- gfx1250 async global->LDS copy (ASYNCcnt path), 32B per lane ----------
// INST_OFFSET is added to BOTH the LDS and global addresses, so one base pair
// covers both 16B halves.

__device__ __forceinline__ void async_b128_x2(unsigned lds_addr,
                                              unsigned long long gaddr) {
  asm volatile("global_load_async_to_lds_b128 %0, %1, off\n\t"
               "global_load_async_to_lds_b128 %0, %1, off offset:16"
               :: "v"(lds_addr), "v"(gaddr)
               : "memory");
}

#define WAIT_ASYNC_4()  asm volatile("s_wait_asynccnt 0x4" ::: "memory")
#define WAIT_ASYNC_0()  asm volatile("s_wait_asynccnt 0x0" ::: "memory")

// ---- Kernel 1: convert x -> f16, weights -> transposed f16 -----------------

__global__ void convert_kernel(const float* __restrict__ x,
                               const float* __restrict__ Wq,
                               const float* __restrict__ Wk,
                               const float* __restrict__ Wv,
                               const float* __restrict__ Wo,
                               _Float16* __restrict__ xh,
                               _Float16* __restrict__ Wqt,
                               _Float16* __restrict__ Wkt,
                               _Float16* __restrict__ Wvt,
                               _Float16* __restrict__ Wot) {
  int i = blockIdx.x * blockDim.x + threadIdx.x;       // 0 .. 1M-1
  if (i >= DD * DD) return;
  int k = i >> 10, n = i & (DD - 1);
  size_t t = (size_t)n * DD + k;                       // Wt[n,k] = W[k,n]
  Wqt[t] = (_Float16)Wq[i];
  Wkt[t] = (_Float16)Wk[i];
  Wvt[t] = (_Float16)Wv[i];
  Wot[t] = (_Float16)Wo[i];
#pragma unroll
  for (int j = 0; j < 4; ++j)
    xh[i + (size_t)j * DD * DD] = (_Float16)x[i + (size_t)j * DD * DD];
}

// ---- WMMA GEMM  out = A[M,K] @ Bt[N,K]^T + bias ----------------------------
// One wave computes a 32(M) x 64(N) tile: 2 A-frags x 4 B-frags -> 8 WMMA per
// 32-slab. k-loop unrolled by 64 with two statically-indexed fragment sets so
// slab s+1's loads issue under slab s's WMMAs with no register rotation.
// OUT_MODE: 0 = f32 [M,N]; 1 = f16 [M,N]; 2 = f16 V^T [b][h][dh][s]

template <int OUT_MODE>
__global__ void gemm_wmma(const _Float16* __restrict__ A,
                          const _Float16* __restrict__ Bt,
                          const float* __restrict__ bias,
                          void* __restrict__ out,
                          int M, int N, int K) {
  int lane = threadIdx.x & 31;
  int wid  = threadIdx.x >> 5;
  int m0 = blockIdx.x * 32;
  int n0 = blockIdx.y * 256 + wid * 64;
  if (m0 >= M || n0 >= N) return;

  v8f  acc[2][4] = {};
  v16h a0[2], a1[2], bf[2][4];

  auto loadset = [&](int s, int k0) {
    a0[s] = loadA(A + (size_t)m0 * K + k0,        K, lane);
    a1[s] = loadA(A + (size_t)(m0 + 16) * K + k0, K, lane);
#pragma unroll
    for (int c = 0; c < 4; ++c)
      bf[s][c] = loadBT(Bt + (size_t)(n0 + c * 16) * K + k0, K, lane);
  };
  auto mul = [&](int s) {
#pragma unroll
    for (int c = 0; c < 4; ++c) {
      acc[0][c] = wmma_f16(a0[s], bf[s][c], acc[0][c]);
      acc[1][c] = wmma_f16(a1[s], bf[s][c], acc[1][c]);
    }
  };

  loadset(0, 0);
  for (int k0 = 0; k0 < K; k0 += 64) {       // K multiple of 64
    __builtin_prefetch(A  + (size_t)(m0 + lane) * K + k0 + 64, 0, 1);
    __builtin_prefetch(Bt + (size_t)(n0 + lane) * K + k0 + 64, 0, 1);
    __builtin_prefetch(Bt + (size_t)(n0 + 32 + lane) * K + k0 + 64, 0, 1);
    loadset(1, k0 + 32);                     // under set-0 WMMAs
    mul(0);
    if (k0 + 64 < K) loadset(0, k0 + 64);    // under set-1 WMMAs
    mul(1);
  }

  int rbase = (lane & 16) ? 8 : 0;
#pragma unroll
  for (int half = 0; half < 2; ++half) {
#pragma unroll
    for (int c = 0; c < 4; ++c) {
      int n = n0 + c * 16 + (lane & 15);
      float bv = bias[n];
#pragma unroll
      for (int r = 0; r < 8; ++r) {
        int row = m0 + half * 16 + rbase + r;
        float v = acc[half][c][r] + bv;
        if (OUT_MODE == 0) {
          ((float*)out)[(size_t)row * N + n] = v;
        } else if (OUT_MODE == 1) {
          ((_Float16*)out)[(size_t)row * N + n] = (_Float16)v;
        } else {
          // V^T: row -> (b, s), n -> (h, dh); store at [b][h][dh][s]
          int bb = row >> 11, sidx = row & (SS - 1);
          int hh = n >> 6,    dh   = n & (DHH - 1);
          ((_Float16*)out)[(((size_t)bb * HHD + hh) * DHH + dh) * SS + sidx] =
              (_Float16)v;
        }
      }
    }
  }
}

// ---- Flash-style masked attention with WMMA --------------------------------
// Block = 128 threads (4 waves) sharing one (b,h); wave w handles q-tile
// qb = blockIdx.x*4 + w. K tile [32 keys x 64 dh] and V^T tile [64 dh x 32
// keys] staged via async global->LDS copies, double buffered.

__global__ void __launch_bounds__(128)
attn_wmma(const _Float16* __restrict__ qh,
          const _Float16* __restrict__ kh,
          const _Float16* __restrict__ vt,
          const float*    __restrict__ mask,
          _Float16* __restrict__ msgs) {
  __shared__ _Float16 lds_k [2][32 * DHH];  // [key][dh],   2 x 4 KiB
  __shared__ _Float16 lds_vt[2][DHH * 32];  // [dh][key],   2 x 4 KiB
  __shared__ _Float16 lds_p [4][16 * 32];   // per-wave P,  4 KiB

  int tid  = threadIdx.x;
  int lane = tid & 31;
  int wid  = tid >> 5;
  int h = blockIdx.y;
  int b = blockIdx.z;
  int qb = blockIdx.x * 4 + wid;            // 0..127

  // staging roles: K tile = 32 rows x 128B (4 chunks); Vt tile = 64 rows x 64B
  int krow = tid >> 2, kcol = (tid & 3) * 16;   // K: row 0..31, 32B quarter
  int vrow = tid >> 1, vcol = (tid & 1) * 16;   // Vt: row 0..63, 32B half
  const _Float16* vthead = vt + ((size_t)b * HHD + h) * DHH * SS;

  const _Float16* qbase = qh + ((size_t)b * SS + qb * 16) * DD + h * DHH;
  v16h aq0 = loadA(qbase + 0,  DD, lane);
  v16h aq1 = loadA(qbase + 32, DD, lane);

  int rbase = (lane & 16) ? 8 : 0;
  float mq[8];
  bool  mqb[8];
#pragma unroll
  for (int r = 0; r < 8; ++r) {
    mq[r]  = mask[(size_t)b * SS + qb * 16 + rbase + r];
    mqb[r] = (mq[r] != 0.f);
  }

  float mrun[8], lrun[8];
#pragma unroll
  for (int r = 0; r < 8; ++r) { mrun[r] = -3.0e38f; lrun[r] = 0.f; }
  v8f acc[4] = {};
  _Float16* pw = lds_p[wid];

  // issue async copies for key tile i into buffer i&1 (4 asyncs per wave)
  auto issue = [&](int i) {
    int buf = i & 1;
    int kb  = i * 32;
    const _Float16* gk = kh + ((size_t)b * SS + kb + krow) * DD + h * DHH + kcol;
    const _Float16* gv = vthead + (size_t)vrow * SS + kb + vcol;
    unsigned lk = (unsigned)(uintptr_t)(&lds_k [buf][krow * DHH + kcol]);
    unsigned lv = (unsigned)(uintptr_t)(&lds_vt[buf][vrow * 32  + vcol]);
    async_b128_x2(lk, (unsigned long long)(uintptr_t)gk);
    async_b128_x2(lv, (unsigned long long)(uintptr_t)gv);
  };

  issue(0);
  for (int it = 0; it < NT; ++it) {
    int buf = it & 1;
    int kb  = it * 32;
    if (it + 1 < NT) { issue(it + 1); WAIT_ASYNC_4(); }
    else             { WAIT_ASYNC_0(); }
    __syncthreads();                         // tile `it` visible to all waves

    // scores: two 16x16 tiles (keys kb..+15, kb+16..+31); K-dim = DH = 64
    float s[2][8];
#pragma unroll
    for (int t = 0; t < 2; ++t) {
      v8f c = {};
      c = wmma_f16(aq0, loadBT(&lds_k[buf][(t * 16) * DHH + 0],  DHH, lane), c);
      c = wmma_f16(aq1, loadBT(&lds_k[buf][(t * 16) * DHH + 32], DHH, lane), c);
      bool mkb = (mask[(size_t)b * SS + kb + t * 16 + (lane & 15)] != 0.f);
#pragma unroll
      for (int r = 0; r < 8; ++r) {
        // mask is exactly {0,1}: mask-before-scale + additive NEG == select
        s[t][r] = (mqb[r] && mkb) ? c[r] * 0.125f : NEGBIG;
      }
    }

    // online softmax: reductions across the 16-lane half-groups
    float p0[8], p1[8];
#pragma unroll
    for (int r = 0; r < 8; ++r) {
      float rm = fmaxf(s[0][r], s[1][r]);
#pragma unroll
      for (int m = 8; m >= 1; m >>= 1) rm = fmaxf(rm, __shfl_xor(rm, m, 32));
      float mn   = fmaxf(mrun[r], rm);
      float corr = __expf(mrun[r] - mn);
      p0[r] = __expf(s[0][r] - mn);
      p1[r] = __expf(s[1][r] - mn);
      float rs = p0[r] + p1[r];
#pragma unroll
      for (int m = 8; m >= 1; m >>= 1) rs += __shfl_xor(rs, m, 32);
      lrun[r] = lrun[r] * corr + rs;
      mrun[r] = mn;
#pragma unroll
      for (int c2 = 0; c2 < 4; ++c2) acc[c2][r] *= corr;
    }

    // transpose P (C-layout -> A-layout) through this wave's LDS region
    int col = lane & 15;
#pragma unroll
    for (int r = 0; r < 8; ++r) {
      pw[(rbase + r) * 32 + col]      = (_Float16)p0[r];
      pw[(rbase + r) * 32 + col + 16] = (_Float16)p1[r];
    }
    v16h pf = loadA(pw, 32, lane);           // same-wave DS ops are in order

    // PV: B-fragments straight from the V^T tile (contiguous 32B per lane)
#pragma unroll
    for (int c2 = 0; c2 < 4; ++c2) {
      v16h bv = loadBT(&lds_vt[buf][(c2 * 16) * 32], 32, lane);
      acc[c2] = wmma_f16(pf, bv, acc[c2]);
    }
    __syncthreads();                         // all reads done before buf reuse
  }

  // normalize (one divide per row), apply query mask, store msgs as f16
#pragma unroll
  for (int r = 0; r < 8; ++r) {
    float scale = mq[r] / lrun[r];
    int row = qb * 16 + rbase + r;
    _Float16* orow = msgs + ((size_t)b * SS + row) * DD + h * DHH + (lane & 15);
#pragma unroll
    for (int c2 = 0; c2 < 4; ++c2)
      orow[c2 * 16] = (_Float16)(acc[c2][r] * scale);
  }
}

// ---------------------------------------------------------------------------

extern "C" void kernel_launch(void* const* d_in, const int* in_sizes, int n_in,
                              void* d_out, int out_size, void* d_ws, size_t ws_size,
                              hipStream_t stream) {
  const float* x    = (const float*)d_in[0];
  const float* mask = (const float*)d_in[1];
  const float* Wq   = (const float*)d_in[2];
  const float* bq   = (const float*)d_in[3];
  const float* Wk   = (const float*)d_in[4];
  const float* bk   = (const float*)d_in[5];
  const float* Wv   = (const float*)d_in[6];
  const float* bv   = (const float*)d_in[7];
  const float* Wo   = (const float*)d_in[8];
  const float* bo   = (const float*)d_in[9];

  char* ws = (char*)d_ws;
  const size_t MiB = 1u << 20;
  _Float16* xh  = (_Float16*)(ws);               //  8 MiB  x as f16
  _Float16* Wqt = (_Float16*)(ws +  8 * MiB);    //  2 MiB  Wq^T f16
  _Float16* Wkt = (_Float16*)(ws + 10 * MiB);    //  2 MiB
  _Float16* Wvt = (_Float16*)(ws + 12 * MiB);    //  2 MiB
  _Float16* Wot = (_Float16*)(ws + 14 * MiB);    //  2 MiB
  _Float16* qhp = (_Float16*)(ws + 16 * MiB);    //  8 MiB  Q f16 [b,s,h,dh]
  _Float16* khp = (_Float16*)(ws + 24 * MiB);    //  8 MiB  K f16 [b,s,h,dh]
  _Float16* vtp = (_Float16*)(ws + 32 * MiB);    //  8 MiB  V^T f16 [b,h,dh,s]
  _Float16* msp = (_Float16*)(ws + 40 * MiB);    //  8 MiB  msgs f16 (48 total)

  // 1) precision conversion + weight transpose
  convert_kernel<<<(DD * DD + 255) / 256, 256, 0, stream>>>(
      x, Wq, Wk, Wv, Wo, xh, Wqt, Wkt, Wvt, Wot);

  // 2) Q/K/V projections (f16 out; V written transposed per head)
  dim3 gg(MM / 32, DD / 256);
  gemm_wmma<1><<<gg, 128, 0, stream>>>(xh, Wqt, bq, qhp, MM, DD, DD);
  gemm_wmma<1><<<gg, 128, 0, stream>>>(xh, Wkt, bk, khp, MM, DD, DD);
  gemm_wmma<2><<<gg, 128, 0, stream>>>(xh, Wvt, bv, vtp, MM, DD, DD);

  // 3) masked flash attention (4 waves/block share one (b,h) K/V stream)
  attn_wmma<<<dim3(SS / 64, HHD, BB), 128, 0, stream>>>(qhp, khp, vtp, mask, msp);

  // 4) output projection (f32 out into d_out)
  gemm_wmma<0><<<gg, 128, 0, stream>>>(msp, Wot, bo, (float*)d_out, MM, DD, DD);
}
